// DynamicLoRAConv1d_32109175505755
// MI455X (gfx1250) — compile-verified, matrix operand order checked by reading, not run on hardware
//
#include <hip/hip_runtime.h>
#include <hip/hip_bf16.h>

typedef __attribute__((ext_vector_type(16))) __bf16 v16bf;
typedef __attribute__((ext_vector_type(8)))  float  v8f;

#define T_IN   2048
#define T_OUT  1024
#define INC    64
#define OUTC   128
#define KW     5
#define KG     320      // INC*KW, ordered kg = kk*64 + ic
#define NT     128      // output columns per workgroup
#define XROWS  260      // 2*(NT-1)+KW-1+1 = 259, padded
#define XSTR   72       // LDS row stride in bf16 elems (144B, 16B aligned, skewed)
#define NSAMP  256
#define BATCH  32
#define R_LORA 8

union Frag { uint4 q[2]; v16bf v; };

__device__ __forceinline__ unsigned short f2bf(float f) {
    unsigned int u = __float_as_uint(f);
    unsigned int r = u + 0x7FFFu + ((u >> 16) & 1u);   // round-to-nearest-even
    return (unsigned short)(r >> 16);
}

// ---------------- Kernel 1: W_eff[b][oc][kg] = conv_w + scale * (B@A), bf16 ----------------
__global__ void __launch_bounds__(256)
weff_kernel(const float* __restrict__ A_flat, const float* __restrict__ B_flat,
            const float* __restrict__ conv_w, const float* __restrict__ lora_scale,
            unsigned short* __restrict__ Weff)
{
    int b   = blockIdx.y;
    int idx = blockIdx.x * 256 + threadIdx.x;          // over OUTC*KG = 40960
    if (idx >= OUTC * KG) return;
    int oc = idx / KG;
    int kg = idx - oc * KG;
    int kk = kg >> 6;          // kg = kk*64 + ic
    int ic = kg & 63;
    int m  = ic * KW + kk;     // A's flat (in_c*k) index

    const float* Ab = A_flat + (size_t)b * (R_LORA * KG);
    const float* Bb = B_flat + (size_t)b * (OUTC * R_LORA) + oc * R_LORA;
    float s = 0.f;
#pragma unroll
    for (int r = 0; r < R_LORA; ++r) s += Bb[r] * Ab[r * KG + m];
    float w = conv_w[(oc * INC + ic) * KW + kk] + lora_scale[0] * s;
    Weff[((size_t)b * OUTC + oc) * KG + kg] = f2bf(w);
}

// ---------------- Kernel 2: implicit-GEMM conv, bias+ReLU, writes pre-norm y ----------------
__global__ void __launch_bounds__(256)
gemm_kernel(const float* __restrict__ x, const unsigned short* __restrict__ Weff,
            const float* __restrict__ conv_b, float* __restrict__ y)
{
    __shared__ __align__(16) unsigned short sX[XROWS * XSTR];   // 37,440 B

    const int tileN = blockIdx.x;          // 0..7
    const int n     = blockIdx.y;          // 0..255
    const int b     = n >> 3;              // batch sample (8 sensors share W_eff)
    const int t0    = tileN * NT;
    const int tin0  = 2 * t0 - 2;          // input t for local j=0 (padding=2)

    // ---- stage x tile transposed to LDS: sX[j][ic] = bf16(x[n][ic][tin0+j]) ----
    const float* xb = x + (size_t)n * INC * T_IN;
    for (int idx = threadIdx.x; idx < INC * XROWS; idx += 256) {
        int ic = idx / XROWS;
        int j  = idx - ic * XROWS;
        int t  = tin0 + j;
        float v = (t >= 0 && t < T_IN && j < 259) ? xb[(size_t)ic * T_IN + t] : 0.f;
        sX[j * XSTR + ic] = f2bf(v);
    }
    __syncthreads();

    const int wave = threadIdx.x >> 5;     // 0..7 -> M rows [16w,16w+16)
    const int lane = threadIdx.x & 31;
    const int half = lane >> 4;            // K-half select per ISA A/B layout
    const int l15  = lane & 15;
    const int mb   = wave * 16;

    v8f acc[8] = {};                       // 8 N-tiles of 16x16 f32

    // per-lane base into W_eff (bf16 elems): row = b*128 + mb + l15
    const size_t aoff = ((size_t)b * OUTC + mb + l15) * KG + half * 8;

#pragma unroll
    for (int kb = 0; kb < KG; kb += 32) {
        const int kk  = kb >> 6;           // kernel tap for this K-chunk
        const int icb = kb & 63;           // in-channel base

        Frag a;
        const uint4* ap = (const uint4*)(Weff + aoff + kb);
        a.q[0] = ap[0];                    // K {0..7} / {8..15}
        a.q[1] = ap[2];                    // K {16..23} / {24..31}

#pragma unroll
        for (int nb = 0; nb < 8; ++nb) {
            const int j = 2 * (nb * 16 + l15) + kk;     // strided-conv column
            Frag bf;
            const uint4* bp = (const uint4*)&sX[j * XSTR + icb + half * 8];
            bf.q[0] = bp[0];
            bf.q[1] = bp[2];
            acc[nb] = __builtin_amdgcn_wmma_f32_16x16x32_bf16(
                false, a.v, false, bf.v, (short)0, acc[nb], false, false);
        }
    }

    // ---- epilogue: bias + ReLU, C-layout (lane<16: M=e, lane>=16: M=e+8) ----
    const float* bp0 = conv_b + mb + half * 8;
    float4 bA = *(const float4*)(bp0);
    float4 bB = *(const float4*)(bp0 + 4);
    float bias[8] = {bA.x, bA.y, bA.z, bA.w, bB.x, bB.y, bB.z, bB.w};

#pragma unroll
    for (int nb = 0; nb < 8; ++nb) {
        const int t = t0 + nb * 16 + l15;
#pragma unroll
        for (int e = 0; e < 8; ++e) {
            const int oc = mb + half * 8 + e;
            float v = acc[nb][e] + bias[e];
            y[((size_t)n * OUTC + oc) * T_OUT + t] = fmaxf(v, 0.f);
        }
    }
}

// ---------------- Kernel 3: per-(sample,group) mean / rsqrt(var+eps) ----------------
__global__ void __launch_bounds__(256)
stats_kernel(const float* __restrict__ y, float* __restrict__ stats)
{
    const int n = blockIdx.x >> 2;
    const int g = blockIdx.x & 3;
    const float* p = y + ((size_t)n * OUTC + g * 32) * T_OUT;   // 32 contiguous channels

    float s = 0.f, ss = 0.f;
    for (int i = threadIdx.x; i < 32 * T_OUT; i += 256) {
        float v = p[i];
        s += v; ss += v * v;
    }
    __shared__ float rs[256], rq[256];
    rs[threadIdx.x] = s; rq[threadIdx.x] = ss;
    __syncthreads();
    for (int off = 128; off > 0; off >>= 1) {
        if (threadIdx.x < off) {
            rs[threadIdx.x] += rs[threadIdx.x + off];
            rq[threadIdx.x] += rq[threadIdx.x + off];
        }
        __syncthreads();
    }
    if (threadIdx.x == 0) {
        const float inv = 1.f / (32.f * T_OUT);
        float mean = rs[0] * inv;
        float var  = rq[0] * inv - mean * mean;
        stats[blockIdx.x * 2 + 0] = mean;
        stats[blockIdx.x * 2 + 1] = rsqrtf(var + 1e-5f);
    }
}

// ---------------- Kernel 4: in-place GroupNorm apply (float4) ----------------
__global__ void __launch_bounds__(256)
norm_kernel(float* __restrict__ y, const float* __restrict__ stats,
            const float* __restrict__ gamma, const float* __restrict__ beta)
{
    const size_t i  = (size_t)blockIdx.x * blockDim.x + threadIdx.x;  // float4 index
    const size_t t4 = i * 4;
    const int c = (int)((t4 >> 10) & (OUTC - 1));
    const int n = (int)(t4 >> 17);           // / (128*1024)
    const int g = c >> 5;
    const float mean = stats[(n * 4 + g) * 2 + 0];
    const float rstd = stats[(n * 4 + g) * 2 + 1];
    const float sc = rstd * gamma[c];
    const float sh = beta[c] - mean * sc;

    float4 v = ((const float4*)y)[i];
    v.x = v.x * sc + sh; v.y = v.y * sc + sh;
    v.z = v.z * sc + sh; v.w = v.w * sc + sh;
    ((float4*)y)[i] = v;
}

extern "C" void kernel_launch(void* const* d_in, const int* in_sizes, int n_in,
                              void* d_out, int out_size, void* d_ws, size_t ws_size,
                              hipStream_t stream)
{
    const float* x        = (const float*)d_in[0];
    const float* A_flat   = (const float*)d_in[1];
    const float* B_flat   = (const float*)d_in[2];
    const float* conv_w   = (const float*)d_in[3];
    const float* conv_b   = (const float*)d_in[4];
    const float* gamma    = (const float*)d_in[5];
    const float* beta     = (const float*)d_in[6];
    // d_in[7]=num_sensors, d_in[8]=r (shapes fixed; hardcoded)
    const float* lora_sc  = (const float*)d_in[9];

    float* y = (float*)d_out;

    unsigned short* Weff = (unsigned short*)d_ws;                 // 32*128*320*2 = 2,621,440 B
    float* stats = (float*)((char*)d_ws + ((size_t)BATCH * OUTC * KG * 2 + 255 & ~(size_t)255));

    // 1) fold LoRA into per-sample bf16 weights
    weff_kernel<<<dim3((OUTC * KG + 255) / 256, BATCH), 256, 0, stream>>>(
        A_flat, B_flat, conv_w, lora_sc, Weff);

    // 2) implicit-GEMM conv + bias + ReLU  (8 t-tiles x 256 samples)
    gemm_kernel<<<dim3(T_OUT / NT, NSAMP), 256, 0, stream>>>(x, Weff, conv_b, y);

    // 3) GroupNorm statistics (256 samples x 4 groups)
    stats_kernel<<<NSAMP * 4, 256, 0, stream>>>(y, stats);

    // 4) apply normalization in place
    const size_t n4 = (size_t)NSAMP * OUTC * T_OUT / 4;
    norm_kernel<<<(unsigned)(n4 / 256), 256, 0, stream>>>(y, stats, gamma, beta);
}
